// ScatterFormer_80049600463069
// MI455X (gfx1250) — compile-verified, matrix operand order checked by reading, not run on hardware
//
#include <hip/hip_runtime.h>
#include <hip/hip_bf16.h>

// ---------------- constants from the reference ----------------
constexpr int NN    = 100000;     // voxels
constexpr int D     = 128;
constexpr int H     = 8;
constexpr int HD    = 16;
constexpr int DFF   = 256;
constexpr int WSZ   = 12;
constexpr int BATCH = 2;
constexpr int SS    = 468;
constexpr int NW    = (SS + WSZ - 1) / WSZ;          // 39
constexpr int SEG   = BATCH * NW * NW;               // 3042
constexpr int GC    = D / 4;                         // 32
constexpr int KS    = WSZ + 1;                       // 13
constexpr int PADC  = KS / 2;                        // 6
constexpr float EPSA = 1e-6f;                        // attention / l2norm eps
constexpr float EPSL = 1e-5f;                        // layernorm eps

typedef __attribute__((ext_vector_type(16))) _Float16 v16h;
typedef __attribute__((ext_vector_type(8)))  _Float16 v8h;
typedef __attribute__((ext_vector_type(8)))  float    v8f;

static inline int nb(long long n, int t) { return (int)((n + t - 1) / t); }

// ---------------- small elementwise kernels ----------------
__global__ void k_fill_f32(float* __restrict__ p, float v, int n) {
  int i = blockIdx.x * blockDim.x + threadIdx.x;
  if (i < n) p[i] = v;
}

__global__ void k_fill_i32(int* __restrict__ p, int v, int n) {
  int i = blockIdx.x * blockDim.x + threadIdx.x;
  if (i < n) p[i] = v;
}

// Swizzle f32 weight W[K x NC] into f16 fragment-ready layout:
// Ws[ct][kc][lane][e], ct = col tile (16 cols), kc = 32-deep K chunk.
// Element (lane, e) of tile (ct, kc): col = ct*16 + (lane&15),
// kk = kc*32 + (e&7) + (e>=8)*16 + (lane>=16)*8   (ISA 7.12.2 16-bit layout)
__global__ void k_wswizzle(const float* __restrict__ W, _Float16* __restrict__ Ws,
                           int K, int NC) {
  int i = blockIdx.x * blockDim.x + threadIdx.x;
  if (i >= K * NC) return;
  int e    = i & 15;
  int lane = (i >> 4) & 31;
  int rest = i >> 9;
  int chunks = K >> 5;
  int kc = rest % chunks;
  int ct = rest / chunks;
  int col = ct * 16 + (lane & 15);
  int kk  = kc * 32 + (e & 7) + ((e >> 3) << 4) + ((lane >> 4) << 3);
  Ws[i] = (_Float16)W[kk * NC + col];
}

// seg/pos from coords, coords layout (N,4): [bi, 0, yi, xi]
__global__ void k_posseg(const int* __restrict__ coords, int* __restrict__ pos,
                         int* __restrict__ seg, int n) {
  int i = blockIdx.x * blockDim.x + threadIdx.x;
  if (i >= n) return;
  int bi = coords[i * 4 + 0];
  int yi = coords[i * 4 + 2];
  int xi = coords[i * 4 + 3];
  seg[i] = bi * (NW * NW) + (xi / WSZ) * NW + (yi / WSZ);
  pos[i] = (xi % WSZ) * WSZ + (yi % WSZ);
}

__global__ void k_buildmap(const int* __restrict__ coords, int* __restrict__ map, int n) {
  int i = blockIdx.x * blockDim.x + threadIdx.x;
  if (i >= n) return;
  int bi = coords[i * 4 + 0];
  int yi = coords[i * 4 + 2];
  int xi = coords[i * 4 + 3];
  map[(bi * SS + xi) * SS + yi] = i;   // duplicates: last-writer, matches .at[].set semantics
}

// xpe16 = f16(src + pe[pos]);  src16 = f16(src)
__global__ void k_addpe(const float* __restrict__ src, const float* __restrict__ pe,
                        const int* __restrict__ pos, _Float16* __restrict__ xpe16,
                        _Float16* __restrict__ src16, int total) {
  int i = blockIdx.x * blockDim.x + threadIdx.x;
  if (i >= total) return;
  int row = i >> 7, col = i & 127;
  float s = src[i];
  xpe16[i] = (_Float16)(s + pe[pos[row] * D + col]);
  src16[i] = (_Float16)s;
}

// per-head l2 normalize in place; i indexes (n*H + h), group of HD=16 floats
__global__ void k_l2norm(float* __restrict__ x, int nh) {
  int i = blockIdx.x * blockDim.x + threadIdx.x;
  if (i >= nh) return;
  float* p = x + (size_t)i * HD;
  float s = 0.f;
#pragma unroll
  for (int j = 0; j < HD; ++j) s += p[j] * p[j];
  float r = rsqrtf(s + EPSA);
#pragma unroll
  for (int j = 0; j < HD; ++j) p[j] *= r;
}

// segment accumulation of kv (SEG,H,16,16) and ksum (SEG,H,16) with f32 atomics
__global__ void k_kvacc(const float* __restrict__ kq, const float* __restrict__ vv,
                        const int* __restrict__ seg, float* __restrict__ kv,
                        float* __restrict__ ksum, int total /* N*128 */) {
  int i = blockIdx.x * blockDim.x + threadIdx.x;
  if (i >= total) return;
  int d = i & 15;
  int h = (i >> 4) & 7;
  int n = i >> 7;
  int s = seg[n];
  float khd = kq[i];
  atomicAdd(&ksum[(s * H + h) * HD + d], khd);
  const float* vp = vv + ((size_t)n << 7) + (h << 4);
  float* kvp = kv + (((size_t)(s * H + h) * HD + d) << 4);
#pragma unroll
  for (int e = 0; e < 16; ++e) atomicAdd(&kvp[e], khd * vp[e]);
}

// out[n,h,:] = (q . kv[seg]) / (q . ksum[seg] + eps), written as f16 GEMM input
__global__ void k_attn(const float* __restrict__ q, const float* __restrict__ kv,
                       const float* __restrict__ ksum, const int* __restrict__ seg,
                       _Float16* __restrict__ out16, int nh) {
  int i = blockIdx.x * blockDim.x + threadIdx.x;
  if (i >= nh) return;
  int h = i & 7;
  int n = i >> 3;
  int s = seg[n];
  float qr[HD];
  const float* qp = q + (size_t)i * HD;
#pragma unroll
  for (int d = 0; d < HD; ++d) qr[d] = qp[d];
  const float* ks = ksum + (size_t)(s * H + h) * HD;
  float den = EPSA;
#pragma unroll
  for (int d = 0; d < HD; ++d) den += qr[d] * ks[d];
  float inv = 1.0f / den;
  const float* kvp = kv + ((size_t)(s * H + h) * HD) * HD;
#pragma unroll
  for (int e = 0; e < HD; ++e) {
    float num = 0.f;
#pragma unroll
    for (int d = 0; d < HD; ++d) num += qr[d] * kvp[d * HD + e];
    out16[(size_t)i * HD + e] = (_Float16)(num * inv);
  }
}

// LayerNorm over D=128: out = ln(a (+ b)) * g + bt. One wave (32 lanes) per row.
__global__ void k_ln(const float* __restrict__ a, const float* __restrict__ b,
                     const float* __restrict__ g, const float* __restrict__ bt,
                     float* __restrict__ outf, float* __restrict__ outf2,
                     _Float16* __restrict__ outh, int M) {
  int row = blockIdx.x * 8 + (threadIdx.x >> 5);
  if (row >= M) return;
  int lane = threadIdx.x & 31;
  const float* ap = a + (size_t)row * D;
  const float* bp = b ? b + (size_t)row * D : nullptr;
  float x[4];
  float sum = 0.f;
#pragma unroll
  for (int j = 0; j < 4; ++j) {
    int c = lane + j * 32;
    float v = ap[c] + (bp ? bp[c] : 0.f);
    x[j] = v;
    sum += v;
  }
#pragma unroll
  for (int m = 16; m >= 1; m >>= 1) sum += __shfl_xor(sum, m, 32);
  float mean = sum * (1.0f / D);
  float var = 0.f;
#pragma unroll
  for (int j = 0; j < 4; ++j) {
    float d = x[j] - mean;
    var += d * d;
  }
#pragma unroll
  for (int m = 16; m >= 1; m >>= 1) var += __shfl_xor(var, m, 32);
  float r = rsqrtf(var * (1.0f / D) + EPSL);
#pragma unroll
  for (int j = 0; j < 4; ++j) {
    int c = lane + j * 32;
    float y = (x[j] - mean) * r * g[c] + bt[c];
    if (outf)  outf[(size_t)row * D + c]  = y;
    if (outf2) outf2[(size_t)row * D + c] = y;
    if (outh)  outh[(size_t)row * D + c]  = (_Float16)y;
  }
}

// sparse depthwise convs via index-map gather; writes cat = [x_id, c_hw, c_w, c_h]
__global__ void k_conv(const float* __restrict__ src, const int* __restrict__ coords,
                       const int* __restrict__ map,
                       const float* __restrict__ w_hw, const float* __restrict__ b_hw,
                       const float* __restrict__ w_w,  const float* __restrict__ b_w,
                       const float* __restrict__ w_h,  const float* __restrict__ b_h,
                       float* __restrict__ cat, int total /* N*GC */) {
  int i = blockIdx.x * blockDim.x + threadIdx.x;
  if (i >= total) return;
  int c = i & (GC - 1);
  int n = i / GC;
  int bi = coords[n * 4 + 0];
  int yi = coords[n * 4 + 2];
  int xi = coords[n * 4 + 3];
  const int* mp = map + (size_t)bi * SS * SS;

  float acc_hw = b_hw[c];
#pragma unroll
  for (int ki = 0; ki < 3; ++ki) {
    int xx = xi + ki - 1;
    if ((unsigned)xx >= (unsigned)SS) continue;
#pragma unroll
    for (int kj = 0; kj < 3; ++kj) {
      int yy = yi + kj - 1;
      if ((unsigned)yy >= (unsigned)SS) continue;
      int m = mp[xx * SS + yy];
      if (m >= 0) acc_hw += w_hw[(ki * 3 + kj) * GC + c] * src[(size_t)m * D + GC + c];
    }
  }
  float acc_w = b_w[c];
#pragma unroll
  for (int kj = 0; kj < KS; ++kj) {
    int yy = yi + kj - PADC;
    if ((unsigned)yy >= (unsigned)SS) continue;
    int m = mp[xi * SS + yy];
    if (m >= 0) acc_w += w_w[kj * GC + c] * src[(size_t)m * D + 2 * GC + c];
  }
  float acc_h = b_h[c];
#pragma unroll
  for (int ki = 0; ki < KS; ++ki) {
    int xx = xi + ki - PADC;
    if ((unsigned)xx >= (unsigned)SS) continue;
    int m = mp[xx * SS + yi];
    if (m >= 0) acc_h += w_h[ki * GC + c] * src[(size_t)m * D + 3 * GC + c];
  }
  float* cp = cat + (size_t)n * D;
  cp[c]          = src[(size_t)n * D + c];   // x_id passthrough
  cp[GC + c]     = acc_hw;
  cp[2 * GC + c] = acc_w;
  cp[3 * GC + c] = acc_h;
}

// ---------------- WMMA GEMM: C[M x NC] = A[M x K](f16) @ W[K x NC] + bias ----------------
// No LDS: A fragments are two contiguous 16B global loads per lane (row-major layout
// matches the 16-bit A fragment layout in 8-half runs); B is pre-swizzled into
// fragment-ready order so each B fragment is one contiguous 32B load (L2-resident).
// 8 waves per block; wave w owns rows [blockIdx.x*128 + w*16, +16) x 64 cols (4 tiles).
template <int K, int NC, bool RELU>
__global__ __launch_bounds__(256) void k_gemm(const _Float16* __restrict__ A,
                                              const _Float16* __restrict__ Ws,
                                              const float* __restrict__ bias,
                                              float* __restrict__ Cf,
                                              _Float16* __restrict__ Ch, int M) {
  const int tid  = threadIdx.x;
  const int wave = tid >> 5;
  const int lane = tid & 31;
  const int rowBase = blockIdx.x * 128 + wave * 16;
  const int colBase = blockIdx.y * 64;
  const int ct0 = colBase >> 4;          // first of 4 col tiles
  constexpr int CHUNKS = K >> 5;         // K/32

  v8f acc0 = {}, acc1 = {}, acc2 = {}, acc3 = {};

  // A fragment source: lane L covers row rowBase+(L&15), k groups offset by (L>=16)*8
  int aRow = rowBase + (lane & 15);
  if (aRow >= M) aRow = M - 1;           // clamp: OOB rows only feed masked outputs
  const _Float16* aPtr = A + (size_t)aRow * K + ((lane >> 4) << 3);

#pragma unroll
  for (int kc = 0; kc < CHUNKS; ++kc) {
    v8h alo = *(const v8h*)(aPtr + kc * 32);        // e0..7  : k = kc*32 + khi + 0..7
    v8h ahi = *(const v8h*)(aPtr + kc * 32 + 16);   // e8..15 : k = kc*32 + 16 + khi + 0..7
    v16h a = __builtin_shufflevector(alo, ahi, 0, 1, 2, 3, 4, 5, 6, 7,
                                               8, 9, 10, 11, 12, 13, 14, 15);
    const _Float16* wBase = Ws + ((size_t)lane << 4);
    v16h b0 = *(const v16h*)(wBase + (((size_t)(ct0 + 0) * CHUNKS + kc) << 9));
    v16h b1 = *(const v16h*)(wBase + (((size_t)(ct0 + 1) * CHUNKS + kc) << 9));
    v16h b2 = *(const v16h*)(wBase + (((size_t)(ct0 + 2) * CHUNKS + kc) << 9));
    v16h b3 = *(const v16h*)(wBase + (((size_t)(ct0 + 3) * CHUNKS + kc) << 9));
    acc0 = __builtin_amdgcn_wmma_f32_16x16x32_f16(false, a, false, b0, (short)0, acc0, false, false);
    acc1 = __builtin_amdgcn_wmma_f32_16x16x32_f16(false, a, false, b1, (short)0, acc1, false, false);
    acc2 = __builtin_amdgcn_wmma_f32_16x16x32_f16(false, a, false, b2, (short)0, acc2, false, false);
    acc3 = __builtin_amdgcn_wmma_f32_16x16x32_f16(false, a, false, b3, (short)0, acc3, false, false);
  }

  // epilogue: C/D layout: VGPR r -> M = r (lanes 0..15) / r+8 (lanes 16..31); N = lane&15
  const int rAdd = (lane >> 4) << 3;
#pragma unroll
  for (int t = 0; t < 4; ++t) {
    const v8f* accp = (t == 0) ? &acc0 : (t == 1) ? &acc1 : (t == 2) ? &acc2 : &acc3;
    int col = colBase + t * 16 + (lane & 15);
    float bs = bias[col];
#pragma unroll
    for (int r = 0; r < 8; ++r) {
      int grow = rowBase + r + rAdd;
      if (grow >= M) continue;
      float v = (*accp)[r] + bs;
      if (RELU) v = fmaxf(v, 0.f);
      if (Cf) Cf[(size_t)grow * NC + col] = v;
      if (Ch) Ch[(size_t)grow * NC + col] = (_Float16)v;
    }
  }
}

// ---------------- orchestration ----------------
extern "C" void kernel_launch(void* const* d_in, const int* in_sizes, int n_in,
                              void* d_out, int out_size, void* d_ws, size_t ws_size,
                              hipStream_t stream) {
  (void)in_sizes; (void)n_in; (void)out_size; (void)ws_size;

  // params: d_in[0]=feats, d_in[1..66]=2 blocks x 33 tensors (dict insertion order), d_in[67]=coords
  auto P = [&](int blk, int j) -> const float* {
    return (const float*)d_in[1 + blk * 33 + j];
  };
  const float* feats  = (const float*)d_in[0];
  const int*   coords = (const int*)d_in[1 + 2 * 33];

  // workspace carve-out
  size_t off = 0;
  auto carve = [&](size_t bytes) -> void* {
    void* p = (char*)d_ws + off;
    off += (bytes + 255) & ~(size_t)255;
    return p;
  };
  float*    res  = (float*)carve((size_t)NN * D * 4);
  float*    src  = (float*)carve((size_t)NN * D * 4);
  float*    qb   = (float*)carve((size_t)NN * D * 4);
  float*    kb   = (float*)carve((size_t)NN * D * 4);
  float*    vb   = (float*)carve((size_t)NN * D * 4);
  float*    t0   = (float*)carve((size_t)NN * D * 4);
  _Float16* a16  = (_Float16*)carve((size_t)NN * DFF * 2);
  _Float16* b16  = (_Float16*)carve((size_t)NN * DFF * 2);
  float*    kv   = (float*)carve((size_t)SEG * H * HD * HD * 4);
  float*    ksum = (float*)carve((size_t)SEG * H * HD * 4);
  int*      segb = (int*)carve((size_t)NN * 4);
  int*      posb = (int*)carve((size_t)NN * 4);
  int*      map  = (int*)carve((size_t)BATCH * SS * SS * 4);
  _Float16* w16  = (_Float16*)carve((size_t)196608 * 2);

  _Float16* wq16   = w16;
  _Float16* wk16   = w16 + 16384;
  _Float16* wv16   = w16 + 32768;
  _Float16* wo16   = w16 + 49152;
  _Float16* l1w16  = w16 + 65536;
  _Float16* l2w16  = w16 + 98304;
  _Float16* cl1w16 = w16 + 131072;
  _Float16* cl2w16 = w16 + 163840;

  const int T = 256;

  // init: src = res = feats
  hipMemcpyAsync(src, feats, (size_t)NN * D * 4, hipMemcpyDeviceToDevice, stream);
  hipMemcpyAsync(res, feats, (size_t)NN * D * 4, hipMemcpyDeviceToDevice, stream);

  // seg/pos + index map (coords constant across blocks)
  k_posseg<<<nb(NN, T), T, 0, stream>>>(coords, posb, segb, NN);
  k_fill_i32<<<nb((long long)BATCH * SS * SS, T), T, 0, stream>>>(map, -1, BATCH * SS * SS);
  k_buildmap<<<nb(NN, T), T, 0, stream>>>(coords, map, NN);

  const dim3 g128(nb(NN, 128), D / 64);     // 128-wide outputs
  const dim3 g256(nb(NN, 128), DFF / 64);   // 256-wide outputs

  for (int b = 0; b < 2; ++b) {
    const bool last = (b == 1);
    // swizzle weights into f16 fragment-ready layout
    k_wswizzle<<<nb(16384, T), T, 0, stream>>>(P(b, 1),  wq16,   128, 128);
    k_wswizzle<<<nb(16384, T), T, 0, stream>>>(P(b, 3),  wk16,   128, 128);
    k_wswizzle<<<nb(16384, T), T, 0, stream>>>(P(b, 5),  wv16,   128, 128);
    k_wswizzle<<<nb(16384, T), T, 0, stream>>>(P(b, 7),  wo16,   128, 128);
    k_wswizzle<<<nb(32768, T), T, 0, stream>>>(P(b, 11), l1w16,  128, 256);
    k_wswizzle<<<nb(32768, T), T, 0, stream>>>(P(b, 13), l2w16,  256, 128);
    k_wswizzle<<<nb(32768, T), T, 0, stream>>>(P(b, 25), cl1w16, 128, 256);
    k_wswizzle<<<nb(32768, T), T, 0, stream>>>(P(b, 27), cl2w16, 256, 128);

    // xpe (a16) and src (b16) in f16
    k_addpe<<<nb((long long)NN * D, T), T, 0, stream>>>(src, P(b, 0), posb, a16, b16, NN * D);

    // q / k / v projections (WMMA)
    k_gemm<128, 128, false><<<g128, T, 0, stream>>>(a16, wq16, P(b, 2), qb, nullptr, NN);
    k_gemm<128, 128, false><<<g128, T, 0, stream>>>(a16, wk16, P(b, 4), kb, nullptr, NN);
    k_gemm<128, 128, false><<<g128, T, 0, stream>>>(b16, wv16, P(b, 6), vb, nullptr, NN);

    k_l2norm<<<nb(NN * H, T), T, 0, stream>>>(qb, NN * H);
    k_l2norm<<<nb(NN * H, T), T, 0, stream>>>(kb, NN * H);

    // segment sums
    k_fill_f32<<<nb((long long)SEG * H * HD * HD, T), T, 0, stream>>>(kv, 0.f, SEG * H * HD * HD);
    k_fill_f32<<<nb((long long)SEG * H * HD, T), T, 0, stream>>>(ksum, 0.f, SEG * H * HD);
    k_kvacc<<<nb((long long)NN * D, T), T, 0, stream>>>(kb, vb, segb, kv, ksum, NN * D);

    // attention output -> f16 (a16)
    k_attn<<<nb(NN * H, T), T, 0, stream>>>(qb, kv, ksum, segb, a16, NN * H);

    // wo projection + residual LN (src = ln(src+att); a16 = f16(src))
    k_gemm<128, 128, false><<<g128, T, 0, stream>>>(a16, wo16, P(b, 8), t0, nullptr, NN);
    k_ln<<<nb(NN, 8), T, 0, stream>>>(src, t0, P(b, 9), P(b, 10), src, nullptr, a16, NN);

    // FFN: relu(src@l1) -> b16(f16); @l2 -> t0; src = ln(src+ff)
    k_gemm<128, 256, true ><<<g256, T, 0, stream>>>(a16, l1w16, P(b, 12), nullptr, b16, NN);
    k_gemm<256, 128, false><<<g128, T, 0, stream>>>(b16, l2w16, P(b, 14), t0, nullptr, NN);
    k_ln<<<nb(NN, 8), T, 0, stream>>>(src, t0, P(b, 15), P(b, 16), src, nullptr, nullptr, NN);

    // sparse depthwise convs -> cat (t0); ln(cat) -> a16(f16)
    k_conv<<<nb((long long)NN * GC, T), T, 0, stream>>>(src, coords, map,
        P(b, 17), P(b, 18), P(b, 19), P(b, 20), P(b, 21), P(b, 22), t0, NN * GC);
    k_ln<<<nb(NN, 8), T, 0, stream>>>(t0, nullptr, P(b, 23), P(b, 24), nullptr, nullptr, a16, NN);

    // conv FFN: relu(@cl1) -> b16; @cl2 -> t0; src = ln(src+cwi)
    k_gemm<128, 256, true ><<<g256, T, 0, stream>>>(a16, cl1w16, P(b, 26), nullptr, b16, NN);
    k_gemm<256, 128, false><<<g128, T, 0, stream>>>(b16, cl2w16, P(b, 28), t0, nullptr, NN);
    k_ln<<<nb(NN, 8), T, 0, stream>>>(src, t0, P(b, 29), P(b, 30), src, nullptr, nullptr, NN);

    // outer residual: out = ln(src + res, rn_g, rn_b)
    if (!last) {
      k_ln<<<nb(NN, 8), T, 0, stream>>>(src, res, P(b, 31), P(b, 32), src, res, nullptr, NN);
    } else {
      k_ln<<<nb(NN, 8), T, 0, stream>>>(src, res, P(b, 31), P(b, 32), (float*)d_out, nullptr, nullptr, NN);
    }
  }
}
